// TransformerLayer_1108101563210
// MI455X (gfx1250) — compile-verified, matrix operand order checked by reading, not run on hardware
//
#include <hip/hip_runtime.h>
#include <hip/hip_bf16.h>
#include <math.h>

typedef __attribute__((ext_vector_type(16))) _Float16 v16h;
typedef __attribute__((ext_vector_type(8)))  float    v8f;

#define S_LEN   2048
#define BATCH   2
#define HDIM    1024
#define NHEADS  16
#define NKVH    4
#define HEADD   64
#define NEXP    8
#define FFN_DIM 4096
#define TOK     (S_LEN * BATCH)          // 4096 tokens
#define QKV_O   ((NHEADS + 2 * NKVH) * HEADD)  // 1536
#define LN_EPS  1e-5f

// ---------------------------------------------------------------------------
// WMMA helper: D = A(16x32 f16) x B(32x16 f16) + C(16x16 f32)
// ---------------------------------------------------------------------------
__device__ __forceinline__ v8f wmma_f16(v16h a, v16h b, v8f c) {
    return __builtin_amdgcn_wmma_f32_16x16x32_f16(
        /*neg_a=*/false, a, /*neg_b=*/false, b,
        /*c_mod=*/(short)0, c, /*reuse_a=*/false, /*reuse_b=*/false);
}

// Fragment gather (ISA 7.12.2, 16-bit operand 16x32):
// lane L: row/col = L%16; elems 0..7 -> K=(L>=16?8:0)+e ; elems 8..15 -> K=(L>=16?24:16)+(e-8)
__device__ __forceinline__ v16h load_frag_rowmajor_f16(const _Float16* __restrict__ rowp, int hi) {
    v16h f;
#pragma unroll
    for (int e = 0; e < 8; ++e) f[e] = rowp[hi * 8 + e];
#pragma unroll
    for (int e = 0; e < 8; ++e) f[8 + e] = rowp[16 + hi * 8 + e];
    return f;
}
__device__ __forceinline__ v16h load_frag_rowmajor_f32(const float* __restrict__ rowp, int hi) {
    v16h f;
#pragma unroll
    for (int e = 0; e < 8; ++e) f[e] = (_Float16)rowp[hi * 8 + e];
#pragma unroll
    for (int e = 0; e < 8; ++e) f[8 + e] = (_Float16)rowp[16 + hi * 8 + e];
    return f;
}

__device__ __forceinline__ float gelu_tanh(float x) {
    float x3 = x * x * x;
    return 0.5f * x * (1.0f + tanhf(0.7978845608028654f * (x + 0.044715f * x3)));
}

// Async global->LDS copy of 32 bytes (2 x B128), CDNA5 ASYNCcnt-tracked DMA path.
__device__ __forceinline__ void async_copy_32B(unsigned lds_addr, const float* gaddr) {
    asm volatile(
        "global_load_async_to_lds_b128 %0, %1, off\n\t"
        "global_load_async_to_lds_b128 %0, %1, off offset:16"
        :: "v"(lds_addr), "v"(gaddr) : "memory");
}
__device__ __forceinline__ void wait_async0() {
    asm volatile("s_wait_asynccnt 0" ::: "memory");
}

// ---------------------------------------------------------------------------
// LayerNorm -> f16 activations. One block per token.
// ---------------------------------------------------------------------------
__global__ __launch_bounds__(256)
void ln_f16_kernel(const float* __restrict__ x, const float* __restrict__ g,
                   const float* __restrict__ b, _Float16* __restrict__ y) {
    __shared__ float row[HDIM];
    __shared__ float red[256];
    const int t = blockIdx.x, tid = threadIdx.x;
    const float* xp = x + (size_t)t * HDIM;
    float s = 0.f;
    for (int i = tid; i < HDIM; i += 256) { float v = xp[i]; row[i] = v; s += v; }
    red[tid] = s; __syncthreads();
    for (int o = 128; o > 0; o >>= 1) { if (tid < o) red[tid] += red[tid + o]; __syncthreads(); }
    float mu = red[0] * (1.f / HDIM); __syncthreads();
    s = 0.f;
    for (int i = tid; i < HDIM; i += 256) { float d = row[i] - mu; s += d * d; }
    red[tid] = s; __syncthreads();
    for (int o = 128; o > 0; o >>= 1) { if (tid < o) red[tid] += red[tid + o]; __syncthreads(); }
    float rs = rsqrtf(red[0] * (1.f / HDIM) + LN_EPS);
    for (int i = tid; i < HDIM; i += 256)
        y[(size_t)t * HDIM + i] = (_Float16)((row[i] - mu) * rs * g[i] + b[i]);
}

// ---------------------------------------------------------------------------
// Generic WMMA GEMM:  C[M,N] = A[M,K](f16) * W[N,K]^T(f32->f16)
// Block: 256 threads = 8 waves; block tile 128(M) x 64(N); wave tile 16x64.
// B tile (64 rows x 32 k of fp32, 8KB) staged in LDS once per block via
// double-buffered GLOBAL_LOAD_ASYNC_TO_LDS_B128 (ASYNCcnt), overlapping the
// next stage's DMA with the current stage's WMMAs.
// Epilogue MODE: 0 f16 store | 1 gelu->f16 | 2 +res->f32 | 3 outf += gate[m]*acc
// ---------------------------------------------------------------------------
template <int MODE>
__global__ __launch_bounds__(256)
void wmma_gemm_kernel(const _Float16* __restrict__ A, const float* __restrict__ W,
                      int M, int N, int K,
                      _Float16* __restrict__ outh, float* __restrict__ outf,
                      const float* __restrict__ res,
                      const float* __restrict__ gate, int gstride, int gidx) {
    __shared__ float Bsh[2][64 * 32];   // [buffer][n-local * 32 + k-local]
    const int tid  = threadIdx.x;
    const int wave = tid >> 5;
    const int lane = tid & 31;
    const int nlo  = lane & 15;
    const int hi   = lane >> 4;
    const int m0 = blockIdx.x * 128 + wave * 16;
    const int n0 = blockIdx.y * 64;
    (void)M;

    // staging assignment: thread -> (row srow of 64, 8-float chunk scol)
    const int srow = tid >> 2;            // 0..63
    const int scol = (tid & 3) * 8;       // 0,8,16,24
    const float*   gsrc = W + (size_t)(n0 + srow) * K + scol;
    const unsigned lds0 = (unsigned)(size_t)(&Bsh[0][srow * 32 + scol]);
    const unsigned lds1 = (unsigned)(size_t)(&Bsh[1][srow * 32 + scol]);

    // prefetch stage 0 into buffer 0
    async_copy_32B(lds0, gsrc);

    v8f acc[4];
#pragma unroll
    for (int c = 0; c < 4; ++c)
#pragma unroll
        for (int r = 0; r < 8; ++r) acc[c][r] = 0.f;

    const _Float16* arow = A + (size_t)(m0 + nlo) * K;   // lane's A row (M = lane&15)
    const int steps = K >> 5;
    for (int s = 0; s < steps; ++s) {
        wait_async0();          // own async writes to LDS complete
        __syncthreads();        // all waves' writes complete -> tile s readable
        if (s + 1 < steps)      // overlap next DMA with this stage's math
            async_copy_32B((s & 1) ? lds0 : lds1, gsrc + (size_t)(s + 1) * 32);

        const float* Bcur = &Bsh[s & 1][0];
        v16h af = load_frag_rowmajor_f16(arow + s * 32, hi);
#pragma unroll
        for (int c = 0; c < 4; ++c) {
            v16h bf = load_frag_rowmajor_f32(Bcur + (c * 16 + nlo) * 32, hi);
            acc[c] = wmma_f16(af, bf, acc[c]);
        }
        __syncthreads();        // protect tile s buffer before it is re-filled
    }

    // C layout: lane holds N = lane&15; VGPR r holds M = r + 8*(lane>=16)
#pragma unroll
    for (int c = 0; c < 4; ++c) {
#pragma unroll
        for (int r = 0; r < 8; ++r) {
            const int m = m0 + r + hi * 8;
            const int n = n0 + c * 16 + nlo;
            const size_t idx = (size_t)m * N + n;
            float v = acc[c][r];
            if (MODE == 0) {
                outh[idx] = (_Float16)v;
            } else if (MODE == 1) {
                outh[idx] = (_Float16)gelu_tanh(v);
            } else if (MODE == 2) {
                outf[idx] = v + res[idx];
            } else {
                float gw = gate[(size_t)m * gstride + gidx];
                outf[idx] += gw * v;
            }
        }
    }
}

// ---------------------------------------------------------------------------
// Causal GQA flash attention. One wave per (batch, head, 16-query tile).
// qkv: f16 [TOK, 1536] = [q(16*64) | k(4*64) | v(4*64)], token t = s*BATCH + b.
// out: f16 [TOK, NHEADS*HEADD]
// ---------------------------------------------------------------------------
__global__ __launch_bounds__(256)
void attn_kernel(const _Float16* __restrict__ qkv, _Float16* __restrict__ out) {
    __shared__ _Float16 plds[8][16 * 32];   // per-wave probability scratch
    const int wv   = threadIdx.x >> 5;
    const int lane = threadIdx.x & 31;
    const int nlo  = lane & 15;
    const int hi   = lane >> 4;
    const int gw = blockIdx.x * 8 + wv;
    const int qt = gw & (S_LEN / 16 - 1);        // 0..127
    const int h  = (gw >> 7) & (NHEADS - 1);
    const int b  = gw >> 11;
    const int q0 = qt * 16;
    const int kvh = h >> 2;                       // NHEADS/NKVH = 4
    const size_t ldr = QKV_O;
    const int koff = NHEADS * HEADD + kvh * HEADD;          // 1024 + kvh*64
    const int voff = (NHEADS + NKVH) * HEADD + kvh * HEADD; // 1280 + kvh*64
    const float scale = 0.125f;  // 1/sqrt(64)

    // Q fragments (16 queries x 64 dims -> two 16x32 A-frags), loaded once
    const _Float16* qp = qkv + ((size_t)(q0 + nlo) * BATCH + b) * ldr + h * HEADD;
    v16h qf0 = load_frag_rowmajor_f16(qp, hi);
    v16h qf1 = load_frag_rowmajor_f16(qp + 32, hi);

    float mrun[8], lrun[8];
    v8f oacc[4];
#pragma unroll
    for (int r = 0; r < 8; ++r) { mrun[r] = -3.0e38f; lrun[r] = 0.f; }
#pragma unroll
    for (int c = 0; c < 4; ++c)
#pragma unroll
        for (int r = 0; r < 8; ++r) oacc[c][r] = 0.f;

    for (int j0 = 0; j0 <= q0 + 15; j0 += 32) {
        // ---- scores: two 16x16 tiles over 32 keys (K-dim = 64 -> 2 WMMAs each)
        v8f s0, s1;
#pragma unroll
        for (int r = 0; r < 8; ++r) { s0[r] = 0.f; s1[r] = 0.f; }
        const _Float16* kp0 = qkv + ((size_t)(j0 + nlo) * BATCH + b) * ldr + koff;
        const _Float16* kp1 = kp0 + (size_t)16 * BATCH * ldr;
        s0 = wmma_f16(qf0, load_frag_rowmajor_f16(kp0, hi), s0);
        s0 = wmma_f16(qf1, load_frag_rowmajor_f16(kp0 + 32, hi), s0);
        s1 = wmma_f16(qf0, load_frag_rowmajor_f16(kp1, hi), s1);
        s1 = wmma_f16(qf1, load_frag_rowmajor_f16(kp1 + 32, hi), s1);

        // ---- causal mask + online softmax (rows live at M = r + 8*hi, col = lane&15)
        const int key0 = j0 + nlo;
        const int key1 = j0 + 16 + nlo;
#pragma unroll
        for (int r = 0; r < 8; ++r) {
            const int m = q0 + r + hi * 8;
            float a0 = (key0 <= m) ? s0[r] * scale : -1.0e30f;
            float a1 = (key1 <= m) ? s1[r] * scale : -1.0e30f;
            float rm = fmaxf(a0, a1);
            rm = fmaxf(rm, __shfl_xor(rm, 1, 32));
            rm = fmaxf(rm, __shfl_xor(rm, 2, 32));
            rm = fmaxf(rm, __shfl_xor(rm, 4, 32));
            rm = fmaxf(rm, __shfl_xor(rm, 8, 32));
            float nm = fmaxf(mrun[r], rm);
            float corr = __expf(mrun[r] - nm);
            float p0 = __expf(a0 - nm);
            float p1 = __expf(a1 - nm);
            float rs_ = p0 + p1;
            rs_ += __shfl_xor(rs_, 1, 32);
            rs_ += __shfl_xor(rs_, 2, 32);
            rs_ += __shfl_xor(rs_, 4, 32);
            rs_ += __shfl_xor(rs_, 8, 32);
            lrun[r] = lrun[r] * corr + rs_;
            mrun[r] = nm;
#pragma unroll
            for (int c = 0; c < 4; ++c) oacc[c][r] *= corr;
            plds[wv][(r + hi * 8) * 32 + nlo]      = (_Float16)p0;
            plds[wv][(r + hi * 8) * 32 + 16 + nlo] = (_Float16)p1;
        }
        // same-wave LDS store->load ordering (CDNA5 split counters)
        asm volatile("s_wait_dscnt 0" ::: "memory");

        // ---- reload P in A-fragment layout (16x32)
        v16h pf = load_frag_rowmajor_f16(&plds[wv][nlo * 32], hi);

        // ---- P x V : 4 output 16x16 tiles over d (B-frag gathered from V rows)
#pragma unroll
        for (int c = 0; c < 4; ++c) {
            v16h vf;
#pragma unroll
            for (int e = 0; e < 16; ++e) {
                const int kk = (e < 8) ? (hi * 8 + e) : (16 + hi * 8 + (e - 8));
                vf[e] = qkv[((size_t)(j0 + kk) * BATCH + b) * ldr + voff + c * 16 + nlo];
            }
            oacc[c] = wmma_f16(pf, vf, oacc[c]);
        }
    }

    // ---- normalize and store f16 [t, h*64 + d]
#pragma unroll
    for (int c = 0; c < 4; ++c) {
#pragma unroll
        for (int r = 0; r < 8; ++r) {
            const int m = q0 + r + hi * 8;
            out[((size_t)m * BATCH + b) * (NHEADS * HEADD) + h * HEADD + c * 16 + nlo] =
                (_Float16)(oacc[c][r] / lrun[r]);
        }
    }
}

// ---------------------------------------------------------------------------
// LN2 + router softmax/top2 gates; also seeds d_out with attention residual.
// One block per token.
// ---------------------------------------------------------------------------
__global__ __launch_bounds__(256)
void ln2_router_kernel(const float* __restrict__ hin, const float* __restrict__ g,
                       const float* __restrict__ bb, const float* __restrict__ Wr,
                       _Float16* __restrict__ lnh, float* __restrict__ wts,
                       float* __restrict__ outseed) {
    __shared__ float row[HDIM];
    __shared__ float red[256];
    __shared__ float lg[NEXP];
    const int t = blockIdx.x, tid = threadIdx.x;
    const float* xp = hin + (size_t)t * HDIM;
    float s = 0.f;
    for (int i = tid; i < HDIM; i += 256) {
        float v = xp[i]; row[i] = v; s += v;
        outseed[(size_t)t * HDIM + i] = v;     // residual seed for MoE accumulate
    }
    red[tid] = s; __syncthreads();
    for (int o = 128; o > 0; o >>= 1) { if (tid < o) red[tid] += red[tid + o]; __syncthreads(); }
    float mu = red[0] * (1.f / HDIM); __syncthreads();
    s = 0.f;
    for (int i = tid; i < HDIM; i += 256) { float d = row[i] - mu; s += d * d; }
    red[tid] = s; __syncthreads();
    for (int o = 128; o > 0; o >>= 1) { if (tid < o) red[tid] += red[tid + o]; __syncthreads(); }
    float rs = rsqrtf(red[0] * (1.f / HDIM) + LN_EPS); __syncthreads();
    for (int i = tid; i < HDIM; i += 256) {
        float v = (row[i] - mu) * rs * g[i] + bb[i];
        row[i] = v;
        lnh[(size_t)t * HDIM + i] = (_Float16)v;
    }
    __syncthreads();
    for (int e = 0; e < NEXP; ++e) {
        float p = 0.f;
        const float* wr = Wr + (size_t)e * HDIM;
        for (int i = tid; i < HDIM; i += 256) p += row[i] * wr[i];
        red[tid] = p; __syncthreads();
        for (int o = 128; o > 0; o >>= 1) { if (tid < o) red[tid] += red[tid + o]; __syncthreads(); }
        if (tid == 0) lg[e] = red[0];
        __syncthreads();
    }
    if (tid == 0) {
        float mx = lg[0];
        for (int e = 1; e < NEXP; ++e) mx = fmaxf(mx, lg[e]);
        float p[NEXP], den = 0.f;
        for (int e = 0; e < NEXP; ++e) { p[e] = __expf(lg[e] - mx); den += p[e]; }
        float inv = 1.f / den;
        for (int e = 0; e < NEXP; ++e) p[e] *= inv;
        int i1 = 0;
        for (int e = 1; e < NEXP; ++e) if (p[e] > p[i1]) i1 = e;
        int i2 = (i1 == 0) ? 1 : 0;
        for (int e = 0; e < NEXP; ++e) if (e != i1 && p[e] > p[i2]) i2 = e;
        for (int e = 0; e < NEXP; ++e)
            wts[(size_t)t * NEXP + e] = (e == i1 || e == i2) ? p[e] : 0.f;
    }
}

// ---------------------------------------------------------------------------
// Orchestration
// ---------------------------------------------------------------------------
extern "C" void kernel_launch(void* const* d_in, const int* in_sizes, int n_in,
                              void* d_out, int out_size, void* d_ws, size_t ws_size,
                              hipStream_t stream) {
    (void)in_sizes; (void)n_in; (void)out_size; (void)ws_size;
    const float* hs    = (const float*)d_in[0];
    const float* ln1w  = (const float*)d_in[1];
    const float* ln1b  = (const float*)d_in[2];
    const float* ln2w  = (const float*)d_in[3];
    const float* ln2b  = (const float*)d_in[4];
    const float* qkvw  = (const float*)d_in[5];
    const float* projw = (const float*)d_in[6];
    const float* routw = (const float*)d_in[7];
    const float* moew1 = (const float*)d_in[8];
    const float* moew2 = (const float*)d_in[9];
    float* out = (float*)d_out;

    char* w = (char*)d_ws;
    _Float16* ln1h  = (_Float16*)w; w += (size_t)TOK * HDIM * 2;
    _Float16* qkvh  = (_Float16*)w; w += (size_t)TOK * QKV_O * 2;
    _Float16* attnh = (_Float16*)w; w += (size_t)TOK * NHEADS * HEADD * 2;
    float*    hattn = (float*)w;    w += (size_t)TOK * HDIM * 4;
    _Float16* ln2h  = (_Float16*)w; w += (size_t)TOK * HDIM * 2;
    float*    gates = (float*)w;    w += (size_t)TOK * NEXP * 4;
    _Float16* h1h   = (_Float16*)w; w += (size_t)TOK * FFN_DIM * 2;

    // 1) LN1 -> f16
    ln_f16_kernel<<<TOK, 256, 0, stream>>>(hs, ln1w, ln1b, ln1h);

    // 2) QKV projection: [TOK,1024] x [1536,1024]^T -> f16 [TOK,1536]
    wmma_gemm_kernel<0><<<dim3(TOK / 128, QKV_O / 64), 256, 0, stream>>>(
        ln1h, qkvw, TOK, QKV_O, HDIM, qkvh, nullptr, nullptr, nullptr, 0, 0);

    // 3) causal GQA flash attention -> f16 [TOK,1024]
    attn_kernel<<<(BATCH * NHEADS * (S_LEN / 16)) / 8, 256, 0, stream>>>(qkvh, attnh);

    // 4) out-projection + residual -> f32 hidden_after_attn
    wmma_gemm_kernel<2><<<dim3(TOK / 128, HDIM / 64), 256, 0, stream>>>(
        attnh, projw, TOK, HDIM, NHEADS * HEADD, nullptr, hattn, hs, nullptr, 0, 0);

    // 5) LN2 + router gates; seed d_out with hidden_after_attn
    ln2_router_kernel<<<TOK, 256, 0, stream>>>(hattn, ln2w, ln2b, routw, ln2h, gates, out);

    // 6) MoE: all 8 experts dense (matches reference), gated accumulate into d_out
    for (int e = 0; e < NEXP; ++e) {
        wmma_gemm_kernel<1><<<dim3(TOK / 128, FFN_DIM / 64), 256, 0, stream>>>(
            ln2h, moew1 + (size_t)e * FFN_DIM * HDIM, TOK, FFN_DIM, HDIM,
            h1h, nullptr, nullptr, nullptr, 0, 0);
        wmma_gemm_kernel<3><<<dim3(TOK / 128, HDIM / 64), 256, 0, stream>>>(
            h1h, moew2 + (size_t)e * HDIM * FFN_DIM, TOK, HDIM, FFN_DIM,
            nullptr, out, nullptr, gates, NEXP, e);
    }
}
